// NonLocalLayer_14620068675634
// MI455X (gfx1250) — matrix operand order repaired and checked by reading, not compiled, not run on hardware
//
#include <hip/hip_runtime.h>
#include <hip/hip_bf16.h>

// ---------------------------------------------------------------------------
// NonLocal attention block for MI455X (gfx1250, wave32, WMMA bf16).
// GEMMs via v_wmma_f32_16x16x32_bf16 (f32 accumulate), double-buffered LDS.
// bf16-source tiles staged with GLOBAL_LOAD_ASYNC_TO_LDS_B128 (ASYNCcnt);
// f32-source tiles staged via b128 loads + v_cvt_pk_bf16_f32 + b128 stores.
// Padded LDS strides kill bank conflicts. Softmax/LayerNorm in f32.
// ---------------------------------------------------------------------------

typedef __attribute__((ext_vector_type(16))) __bf16 v16bf;
typedef __attribute__((ext_vector_type(8)))  __bf16 v8bf;
typedef __attribute__((ext_vector_type(8)))  float  v8f;

#define N_BATCH 32
#define C_ST_   2048
#define C_LT_   2048
#define LAT_    512
#define S_POS   512
#define L_POS   2048

// Block tile 128x128, K-step 32 (one WMMA K). 256 threads = 8 waves (4M x 2N),
// each wave owns 32(M) x 64(N) = 2x4 WMMA tiles.
#define BM 128
#define BN 128
#define BK 32
#define LDA 40    // BK + 8 pad  (80 B row stride, 16 B multiple)
#define LDB 136   // BN + 8 pad  (272 B row stride -> bank = 17k mod 64, all distinct)

__device__ __forceinline__ void wait_async0() {
#if __has_builtin(__builtin_amdgcn_s_wait_asynccnt)
    __builtin_amdgcn_s_wait_asynccnt(0);
#else
    asm volatile("s_wait_asynccnt 0x0" ::: "memory");
#endif
}

// Per-lane 16-byte async copy global -> LDS (tracked by ASYNCcnt).
__device__ __forceinline__ void async_copy_b128(const void* gptr, void* lptr) {
    // Generic LDS pointer low 32 bits == workgroup-relative LDS byte offset.
    unsigned int loff = (unsigned int)(unsigned long)lptr;
    asm volatile("global_load_async_to_lds_b128 %0, %1, off"
                 :: "v"(loff), "v"(gptr) : "memory");
}

template<bool A_F32, bool B_F32, bool OUT_F32, bool TRANS_C, bool BIAS>
__global__ __launch_bounds__(256)
void wmma_gemm_kernel(const void* __restrict__ Ap, const void* __restrict__ Bp,
                      const float* __restrict__ bias, void* __restrict__ Cp,
                      int M, int K, int P,
                      long sA, long sB, long sC, float scale)
{
    __shared__ __attribute__((aligned(16))) __bf16 ldsA[2][BM * LDA];
    __shared__ __attribute__((aligned(16))) __bf16 ldsB[2][BK * LDB];

    const int tid   = threadIdx.x;
    const int batch = blockIdx.z;
    const int m0    = blockIdx.y * BM;
    const int p0    = blockIdx.x * BN;

    const float*  Af = nullptr; const __bf16* Ab = nullptr;
    const float*  Bf = nullptr; const __bf16* Bb = nullptr;
    if constexpr (A_F32) Af = (const float*)Ap + (long)batch * sA;
    else                 Ab = (const __bf16*)Ap + (long)batch * sA;
    if constexpr (B_F32) Bf = (const float*)Bp + (long)batch * sB;
    else                 Bb = (const __bf16*)Bp + (long)batch * sB;

    const int lane  = tid & 31;
    const int wave  = tid >> 5;
    const int wm    = (wave >> 1) * 32;   // 4 waves in M
    const int wn    = (wave & 1) * 64;    // 2 waves in N
    const int aRow  = lane & 15;
    const int aHalf = lane >> 4;
    const int nCol  = lane & 15;
    const int mHi   = (lane >> 4) * 8;

    // register staging for f32 sources (compiled out otherwise)
    float4 fA[4];
    float4 fB[4];

    // ---- staging: A ----
    auto stageA = [&](int k0, int buf) {
        if constexpr (A_F32) {
            (void)buf; // fetch only; store via storeA
#pragma unroll
            for (int i = 0; i < 2; ++i) {
                int s_ = i * 256 + tid; int r = s_ >> 2, c8 = s_ & 3;
                const float* gp = Af + (long)(m0 + r) * K + k0 + c8 * 8;
                fA[2 * i]     = *(const float4*)(gp);
                fA[2 * i + 1] = *(const float4*)(gp + 4);
            }
        } else {
#pragma unroll
            for (int i = 0; i < 2; ++i) {
                int s_ = i * 256 + tid; int r = s_ >> 2, c8 = s_ & 3;
                async_copy_b128(Ab + (long)(m0 + r) * K + k0 + c8 * 8,
                                &ldsA[buf][r * LDA + c8 * 8]);
            }
        }
    };
    auto storeA = [&](int buf) {
        if constexpr (A_F32) {
#pragma unroll
            for (int i = 0; i < 2; ++i) {
                int s_ = i * 256 + tid; int r = s_ >> 2, c8 = s_ & 3;
                const float4 a = fA[2 * i], b = fA[2 * i + 1];
                v8bf t = { (__bf16)a.x, (__bf16)a.y, (__bf16)a.z, (__bf16)a.w,
                           (__bf16)b.x, (__bf16)b.y, (__bf16)b.z, (__bf16)b.w };
                *(v8bf*)&ldsA[buf][r * LDA + c8 * 8] = t;
            }
        }
    };
    // ---- staging: B ----
    auto stageB = [&](int k0, int buf) {
        if constexpr (B_F32) {
            (void)buf;
#pragma unroll
            for (int i = 0; i < 2; ++i) {
                int s_ = i * 256 + tid; int r = s_ >> 4, c8 = s_ & 15;
                const float* gp = Bf + (long)(k0 + r) * P + p0 + c8 * 8;
                fB[2 * i]     = *(const float4*)(gp);
                fB[2 * i + 1] = *(const float4*)(gp + 4);
            }
        } else {
#pragma unroll
            for (int i = 0; i < 2; ++i) {
                int s_ = i * 256 + tid; int r = s_ >> 4, c8 = s_ & 15;
                async_copy_b128(Bb + (long)(k0 + r) * P + p0 + c8 * 8,
                                &ldsB[buf][r * LDB + c8 * 8]);
            }
        }
    };
    auto storeB = [&](int buf) {
        if constexpr (B_F32) {
#pragma unroll
            for (int i = 0; i < 2; ++i) {
                int s_ = i * 256 + tid; int r = s_ >> 4, c8 = s_ & 15;
                const float4 a = fB[2 * i], b = fB[2 * i + 1];
                v8bf t = { (__bf16)a.x, (__bf16)a.y, (__bf16)a.z, (__bf16)a.w,
                           (__bf16)b.x, (__bf16)b.y, (__bf16)b.z, (__bf16)b.w };
                *(v8bf*)&ldsB[buf][r * LDB + c8 * 8] = t;
            }
        }
    };

    constexpr bool ANY_ASYNC = (!A_F32) || (!B_F32);

    v8f acc[2][4] = {};

    // prologue: stage tile 0 into buffer 0
    stageA(0, 0); stageB(0, 0);
    storeA(0);    storeB(0);
    if constexpr (ANY_ASYNC) wait_async0();
    __syncthreads();

    int buf = 0;
    for (int k0 = 0; k0 < K; k0 += BK) {
        const bool more = (k0 + BK) < K;
        if (more) { stageA(k0 + BK, buf ^ 1); stageB(k0 + BK, buf ^ 1); }

        // ---- fragments from LDS (documented CDNA5 layouts) ----
        v16bf aF[2], bF[4];
#pragma unroll
        for (int t = 0; t < 2; ++t) {
            union { v16bf v; v8bf h[2]; } ua;
            const __bf16* baseA = &ldsA[buf][(wm + t * 16 + aRow) * LDA + aHalf * 8];
            ua.h[0] = *(const v8bf*)(baseA);
            ua.h[1] = *(const v8bf*)(baseA + 16);
            aF[t] = ua.v;
        }
#pragma unroll
        for (int t = 0; t < 4; ++t) {
            const __bf16* baseB = &ldsB[buf][wn + t * 16 + nCol];
#pragma unroll
            for (int s = 0; s < 16; ++s) {
                int kk = (s < 8) ? (aHalf * 8 + s) : (16 + aHalf * 8 + (s - 8));
                bF[t][s] = baseB[kk * LDB];
            }
        }
#pragma unroll
        for (int ti = 0; ti < 2; ++ti)
#pragma unroll
            for (int tj = 0; tj < 4; ++tj)
                acc[ti][tj] = __builtin_amdgcn_wmma_f32_16x16x32_bf16(
                    false, aF[ti], false, bF[tj], (short)0, acc[ti][tj],
                    false, false);

        if (more) { storeA(buf ^ 1); storeB(buf ^ 1); }
        if constexpr (ANY_ASYNC) wait_async0();
        __syncthreads();
        buf ^= 1;
    }

    // ---- epilogue ----
    if constexpr (TRANS_C) {
        // write C^T (bf16): 8 consecutive gm per lane -> single b128 store
        __bf16* Cb = (__bf16*)Cp + (long)batch * sC;
#pragma unroll
        for (int ti = 0; ti < 2; ++ti)
#pragma unroll
            for (int tj = 0; tj < 4; ++tj) {
                int gmBase = m0 + wm + ti * 16 + mHi;
                int gn     = p0 + wn + tj * 16 + nCol;
                v8bf t;
#pragma unroll
                for (int r = 0; r < 8; ++r) {
                    float v = acc[ti][tj][r] * scale;
                    if constexpr (BIAS) v += bias[gmBase + r];
                    t[r] = (__bf16)v;
                }
                *(v8bf*)&Cb[(long)gn * M + gmBase] = t;
            }
    } else {
        float*  Cf = nullptr; __bf16* Cb = nullptr;
        if constexpr (OUT_F32) Cf = (float*)Cp + (long)batch * sC;
        else                   Cb = (__bf16*)Cp + (long)batch * sC;
#pragma unroll
        for (int ti = 0; ti < 2; ++ti)
#pragma unroll
            for (int tj = 0; tj < 4; ++tj)
#pragma unroll
                for (int r = 0; r < 8; ++r) {
                    int gm = m0 + wm + ti * 16 + mHi + r;
                    int gn = p0 + wn + tj * 16 + nCol;
                    float v = acc[ti][tj][r] * scale;
                    if constexpr (BIAS) v += bias[gm];
                    if constexpr (OUT_F32) Cf[(long)gm * P + gn] = v;
                    else                   Cb[(long)gm * P + gn] = (__bf16)v;
                }
    }
}

// ---------------------------------------------------------------------------
// Row softmax over L=2048; one block per (n,s) row. Reads f32 scores,
// writes TRANSPOSED bf16 probabilities pT[l][s] (WMMA-ready, no trans stage).
// The 64 MB pT fits in the 192 MB L2, so scattered b16 stores coalesce there.
// ---------------------------------------------------------------------------
__global__ __launch_bounds__(256)
void softmax_kernel(const float* __restrict__ scores, __bf16* __restrict__ pT)
{
    __shared__ float red[256];
    const long row = blockIdx.x;                 // row = n*S + s
    const int  n   = (int)(row >> 9);
    const int  s   = (int)(row & (S_POS - 1));
    const float* x = scores + row * (long)L_POS;
    __bf16* y = pT + (long)n * L_POS * S_POS + s;
    const int tid = threadIdx.x;

    float v[8];
    float m = -3.4e38f;
#pragma unroll
    for (int i = 0; i < 8; ++i) { v[i] = x[i * 256 + tid]; m = fmaxf(m, v[i]); }
    red[tid] = m; __syncthreads();
    for (int st = 128; st > 0; st >>= 1) {
        if (tid < st) red[tid] = fmaxf(red[tid], red[tid + st]);
        __syncthreads();
    }
    m = red[0]; __syncthreads();

    float sum = 0.0f;
#pragma unroll
    for (int i = 0; i < 8; ++i) { v[i] = __expf(v[i] - m); sum += v[i]; }
    red[tid] = sum; __syncthreads();
    for (int st = 128; st > 0; st >>= 1) {
        if (tid < st) red[tid] += red[tid + st];
        __syncthreads();
    }
    const float inv = 1.0f / red[0];
#pragma unroll
    for (int i = 0; i < 8; ++i)
        y[(long)(i * 256 + tid) * S_POS] = (__bf16)(v[i] * inv);
}

// ---------------------------------------------------------------------------
// LayerNorm over (LAT*S)=262144 elems per batch + affine + ReLU -> bf16.
// ---------------------------------------------------------------------------
__global__ __launch_bounds__(256)
void layernorm_relu_kernel(const float* __restrict__ xin,
                           const float* __restrict__ lnw,
                           const float* __restrict__ lnb,
                           __bf16* __restrict__ y)
{
    __shared__ float rsum[256];
    __shared__ float rsq[256];
    const int n = blockIdx.x;
    const long NN = (long)LAT_ * S_POS;
    const float* x = xin + n * NN;
    __bf16* yo = y + n * NN;
    const int tid = threadIdx.x;

    float s = 0.0f, q = 0.0f;
    for (long i = tid; i < NN; i += 256) { float v = x[i]; s += v; q += v * v; }
    rsum[tid] = s; rsq[tid] = q; __syncthreads();
    for (int st = 128; st > 0; st >>= 1) {
        if (tid < st) { rsum[tid] += rsum[tid + st]; rsq[tid] += rsq[tid + st]; }
        __syncthreads();
    }
    const float mu   = rsum[0] / (float)NN;
    const float var  = rsq[0] / (float)NN - mu * mu;
    const float rstd = rsqrtf(var + 1e-5f);
    for (long i = tid; i < NN; i += 256) {
        float v = (x[i] - mu) * rstd * lnw[i] + lnb[i];
        yo[i] = (__bf16)fmaxf(v, 0.0f);
    }
}

// ---------------------------------------------------------------------------
extern "C" void kernel_launch(void* const* d_in, const int* in_sizes, int n_in,
                              void* d_out, int out_size, void* d_ws, size_t ws_size,
                              hipStream_t stream)
{
    const float* st    = (const float*)d_in[0];
    const float* lt    = (const float*)d_in[1];
    const float* w_st  = (const float*)d_in[2];
    const float* b_st  = (const float*)d_in[3];
    const float* w_lt  = (const float*)d_in[4];
    const float* b_lt  = (const float*)d_in[5];
    const float* w_g   = (const float*)d_in[6];
    const float* b_g   = (const float*)d_in[7];
    const float* ln_w  = (const float*)d_in[8];
    const float* ln_b  = (const float*)d_in[9];
    const float* w_out = (const float*)d_in[10];
    const float* b_out = (const float*)d_in[11];
    float* out = (float*)d_out;

    // Workspace layout (~384 MB)
    char* ws = (char*)d_ws;
    size_t off = 0;
    auto wsalloc = [&](size_t bytes) {
        void* p = ws + off;
        off += (bytes + 255) & ~(size_t)255;
        return p;
    };
    __bf16* thetaT = (__bf16*)wsalloc((size_t)N_BATCH * S_POS * LAT_ * 2); // 16 MB, [s][c]
    __bf16* phi    = (__bf16*)wsalloc((size_t)N_BATCH * LAT_ * L_POS * 2); // 64 MB, [c][l]
    __bf16* gmat   = (__bf16*)wsalloc((size_t)N_BATCH * LAT_ * L_POS * 2); // 64 MB, [c][l]
    float*  sc     = (float*) wsalloc((size_t)N_BATCH * S_POS * L_POS * 4); // 128 MB, [s][l]
    __bf16* pmatT  = (__bf16*)wsalloc((size_t)N_BATCH * L_POS * S_POS * 2); // 64 MB, [l][s]
    float*  attn   = (float*) wsalloc((size_t)N_BATCH * LAT_ * S_POS * 4); // 32 MB, [c][s]
    __bf16* ybf    = (__bf16*)wsalloc((size_t)N_BATCH * LAT_ * S_POS * 2); // 16 MB, [c][s]
    (void)ws_size; (void)in_sizes; (void)n_in; (void)out_size;

    const dim3 blk(256);
    const float inv_sqrt_lat = 0.04419417382415922f; // 1/sqrt(512)

    // thetaT = (w_st @ st + b_st)^T        (M=512, K=2048, P=512)  -> bf16 [s][c]
    wmma_gemm_kernel<true, true, false, true, true>
        <<<dim3(S_POS / BN, LAT_ / BM, N_BATCH), blk, 0, stream>>>(
            w_st, st, b_st, thetaT, LAT_, C_ST_, S_POS,
            0L, (long)C_ST_ * S_POS, (long)S_POS * LAT_, 1.0f);

    // phi = w_lt @ lt + b_lt               (M=512, K=2048, P=2048) -> bf16
    wmma_gemm_kernel<true, true, false, false, true>
        <<<dim3(L_POS / BN, LAT_ / BM, N_BATCH), blk, 0, stream>>>(
            w_lt, lt, b_lt, phi, LAT_, C_LT_, L_POS,
            0L, (long)C_LT_ * L_POS, (long)LAT_ * L_POS, 1.0f);

    // g = w_g @ lt + b_g                   (M=512, K=2048, P=2048) -> bf16
    wmma_gemm_kernel<true, true, false, false, true>
        <<<dim3(L_POS / BN, LAT_ / BM, N_BATCH), blk, 0, stream>>>(
            w_g, lt, b_g, gmat, LAT_, C_LT_, L_POS,
            0L, (long)C_LT_ * L_POS, (long)LAT_ * L_POS, 1.0f);

    // scores = (thetaT @ phi) / sqrt(LAT)  (M=S=512, K=LAT=512, P=L=2048) -> f32
    wmma_gemm_kernel<false, false, true, false, false>
        <<<dim3(L_POS / BN, S_POS / BM, N_BATCH), blk, 0, stream>>>(
            thetaT, phi, nullptr, sc, S_POS, LAT_, L_POS,
            (long)S_POS * LAT_, (long)LAT_ * L_POS, (long)S_POS * L_POS,
            inv_sqrt_lat);

    // pT = softmax_L(scores)^T -> bf16 [l][s]
    softmax_kernel<<<N_BATCH * S_POS, blk, 0, stream>>>(sc, pmatT);

    // attn = g @ pT                        (M=LAT=512, K=L=2048, P=S=512) -> f32
    wmma_gemm_kernel<false, false, true, false, false>
        <<<dim3(S_POS / BN, LAT_ / BM, N_BATCH), blk, 0, stream>>>(
            gmat, pmatT, nullptr, attn, LAT_, L_POS, S_POS,
            (long)LAT_ * L_POS, (long)L_POS * S_POS, (long)LAT_ * S_POS, 1.0f);

    // y = relu(LN(attn) * ln_w + ln_b) -> bf16
    layernorm_relu_kernel<<<N_BATCH, blk, 0, stream>>>(attn, ln_w, ln_b, ybf);

    // out = w_out @ y + b_out              (M=2048, K=512, P=512) -> f32 (d_out)
    wmma_gemm_kernel<true, false, true, false, true>
        <<<dim3(S_POS / BN, C_ST_ / BM, N_BATCH), blk, 0, stream>>>(
            w_out, ybf, b_out, out, C_ST_, LAT_, S_POS,
            0L, (long)LAT_ * S_POS, (long)C_ST_ * S_POS, 1.0f);
}
